// MultiHeadAttention_89816356094017
// MI455X (gfx1250) — matrix operand order repaired and checked by reading, not compiled, not run on hardware
//
#include <hip/hip_runtime.h>
#include <hip/hip_bf16.h>
#include <math.h>

// ---------------------------------------------------------------------------
// MultiHeadAttention for MI455X (gfx1250, wave32, WMMA)
//   x[4096,768] -> per-head QKV proj -> causal flash attention -> out proj
// All GEMMs: v_wmma_f32_16x16x32_bf16. All fragment loads are b128-wide:
// every B operand is kept K-contiguous (transposed) in memory.
// ---------------------------------------------------------------------------

typedef __bf16 bf16;
typedef bf16  v8bf  __attribute__((ext_vector_type(8)));    // 16 B
typedef bf16  v16bf __attribute__((ext_vector_type(16)));   // 32 B
typedef float v8f   __attribute__((ext_vector_type(8)));

#define T_SEQ 4096
#define DM    768
#define NH    12
#define DH    64

__device__ __forceinline__ v8f wmma_bf16(v16bf a, v16bf b, v8f c) {
  // (neg_a, A, neg_b, B, c_mod, C, reuse_a, reuse_b)
  return __builtin_amdgcn_wmma_f32_16x16x32_bf16(false, a, false, b,
                                                 (short)0, c, false, false);
}

// A fragment: 16(M) x 32(K) bf16 tile, row-major source, row stride ld.
// ISA: lane<16 -> row=lane, K={kb..kb+7, 16+kb..16+kb+7}, kb=(lane>>4)*8.
// Two contiguous 16B loads per lane.
__device__ __forceinline__ v16bf load_a(const bf16* p, int ld) {
  const int lane = threadIdx.x & 31;
  const bf16* r  = p + (lane & 15) * ld + (lane >> 4) * 8;
  v8bf lo = *(const v8bf*)(r);
  v8bf hi = *(const v8bf*)(r + 16);
  return __builtin_shufflevector(lo, hi, 0, 1, 2, 3, 4, 5, 6, 7,
                                 8, 9, 10, 11, 12, 13, 14, 15);
}

// B fragment from a K-contiguous (transposed) source: logical B[k][n] =
// p[n*ld + k].  ISA: lane -> col=lane&15, K={kb..kb+7, 16+kb..16+kb+7}, kb=(lane>>4)*8.
// Two contiguous 16B loads per lane.
__device__ __forceinline__ v16bf load_bT(const bf16* p, int ld) {
  const int lane = threadIdx.x & 31;
  const bf16* r  = p + (lane & 15) * ld + (lane >> 4) * 8;
  v8bf lo = *(const v8bf*)(r);
  v8bf hi = *(const v8bf*)(r + 16);
  return __builtin_shufflevector(lo, hi, 0, 1, 2, 3, 4, 5, 6, 7,
                                 8, 9, 10, 11, 12, 13, 14, 15);
}

// ------------------------- fp32 -> bf16 converters -------------------------
__global__ void cvt_bf16_kernel(const float* __restrict__ s,
                                bf16* __restrict__ d, int n) {
  int i = blockIdx.x * blockDim.x + threadIdx.x;
  if (i < n) d[i] = (bf16)s[i];
}

// wq/wk/wv: [H][C][Dh] fp32 -> [H][Dh][C] bf16 (K-contiguous for WMMA B)
__global__ void cvt_wqkv_T_kernel(const float* __restrict__ s,
                                  bf16* __restrict__ d, int n) {
  int i = blockIdx.x * blockDim.x + threadIdx.x;
  if (i >= n) return;
  int h   = i / (DM * DH);
  int rem = i - h * DM * DH;
  int c   = rem / DH;
  int dd  = rem - c * DH;
  d[((size_t)h * DH + dd) * DM + c] = (bf16)s[i];
}

// wo: [C][O] fp32 -> [O][C] bf16
__global__ void cvt_wo_T_kernel(const float* __restrict__ s,
                                bf16* __restrict__ d, int n) {
  int i = blockIdx.x * blockDim.x + threadIdx.x;
  if (i >= n) return;
  int c = i / DM;
  int o = i - c * DM;
  d[(size_t)o * DM + c] = (bf16)s[i];
}

// ------------------------------ QKV projection -----------------------------
// grid = (T/16, 3*NH), block = 32.  One wave -> 16x64 strip of one head.
// Q,K stored [H][T][Dh]; V stored transposed [H][Dh][T] so the attention
// PV B-fragments are K-contiguous.
__global__ void __launch_bounds__(32)
qkv_proj_kernel(const bf16* __restrict__ xb,
                const bf16* __restrict__ wqT, const bf16* __restrict__ wkT,
                const bf16* __restrict__ wvT,
                const float* __restrict__ bq, const float* __restrict__ bk,
                const float* __restrict__ bv,
                bf16* __restrict__ q, bf16* __restrict__ k,
                bf16* __restrict__ vT) {
  const int t0  = blockIdx.x * 16;
  const int h   = blockIdx.y % NH;
  const int mat = blockIdx.y / NH;   // 0=Q, 1=K, 2=V

  const bf16* wT; const float* bias;
  if (mat == 0)      { wT = wqT; bias = bq; }
  else if (mat == 1) { wT = wkT; bias = bk; }
  else               { wT = wvT; bias = bv; }
  wT += (size_t)h * DH * DM;   // [Dh][C]

  v8f acc[4];
#pragma unroll
  for (int i = 0; i < 4; ++i) acc[i] = (v8f){0.f,0.f,0.f,0.f,0.f,0.f,0.f,0.f};

  for (int kc = 0; kc < DM; kc += 32) {
    v16bf a = load_a(xb + (size_t)t0 * DM + kc, DM);
#pragma unroll
    for (int nt = 0; nt < 4; ++nt)
      acc[nt] = wmma_bf16(a, load_bT(wT + (size_t)(nt * 16) * DM + kc, DM),
                          acc[nt]);
  }

  const int lane = threadIdx.x & 31;
  const int col  = lane & 15;
  const int rb   = (lane >> 4) * 8;
  const float qs = (mat == 0) ? 0.125f : 1.0f;   // 1/sqrt(Dh) folded into Q
#pragma unroll
  for (int nt = 0; nt < 4; ++nt) {
    const int dd = nt * 16 + col;
    const float bval = bias[h * DH + dd];
#pragma unroll
    for (int i = 0; i < 8; ++i) {
      float o = (acc[nt][i] + bval) * qs;
      if (mat == 2)
        vT[((size_t)h * DH + dd) * T_SEQ + (t0 + rb + i)] = (bf16)o;
      else {
        bf16* dst = (mat == 0) ? q : k;
        dst[((size_t)h * T_SEQ + (t0 + rb + i)) * DH + dd] = (bf16)o;
      }
    }
  }
}

// --------------------------- causal flash attention ------------------------
// grid = (T/16, NH), block = 32. One wave owns a 16-query tile of one head.
__global__ void __launch_bounds__(32)
attn_kernel(const bf16* __restrict__ Q, const bf16* __restrict__ K,
            const bf16* __restrict__ VT, bf16* __restrict__ H) {
  const int q0   = blockIdx.x * 16;
  const int h    = blockIdx.y;
  const int lane = threadIdx.x & 31;
  const int col  = lane & 15;
  const int rb   = (lane >> 4) * 8;

  const bf16* Qh  = Q  + (size_t)h * T_SEQ * DH;   // [T][Dh]
  const bf16* Kh  = K  + (size_t)h * T_SEQ * DH;   // [T][Dh]
  const bf16* VTh = VT + (size_t)h * DH * T_SEQ;   // [Dh][T]

  __shared__ float sf[16][33];                              // f32 score staging
  __shared__ __attribute__((aligned(16))) bf16 pbuf[16][32]; // P (A-fragment)
  __shared__ float mst[16], lst[16], scs[16];

  // Q fragments resident for the whole key loop (Dh=64 -> 2 K-chunks).
  v16bf qa0 = load_a(Qh + (size_t)q0 * DH, DH);
  v16bf qa1 = load_a(Qh + (size_t)q0 * DH + 32, DH);

  v8f acc[4];
#pragma unroll
  for (int i = 0; i < 4; ++i) acc[i] = (v8f){0.f,0.f,0.f,0.f,0.f,0.f,0.f,0.f};

  if (lane < 16) { mst[lane] = -3.0e38f; lst[lane] = 0.f; }
  __syncthreads();

  const int qlast = q0 + 15;
  for (int kb = 0; kb <= qlast; kb += 32) {
    if (kb + 32 <= qlast)   // prefetch next K tile (global_prefetch_b8)
      __builtin_prefetch(Kh + (size_t)(kb + 32) * DH + lane * 4, 0, 1);

    // S[16x32] = Q[16x64] * K[kb..kb+31]^T  (B[k][n]=K[n][k]: K-contiguous)
#pragma unroll
    for (int nt = 0; nt < 2; ++nt) {
      v8f s = (v8f){0.f,0.f,0.f,0.f,0.f,0.f,0.f,0.f};
      s = wmma_bf16(qa0, load_bT(Kh + (size_t)(kb + nt * 16) * DH, DH), s);
      s = wmma_bf16(qa1, load_bT(Kh + (size_t)(kb + nt * 16) * DH + 32, DH), s);
#pragma unroll
      for (int i = 0; i < 8; ++i) sf[rb + i][nt * 16 + col] = s[i];
    }
    __syncthreads();

    // Online softmax, one query row per lane 0..15 (causal mask here).
    if (lane < 16) {
      const int r = lane, qg = q0 + r;
      float mo = mst[r];
      float rm = -3.0e38f;
#pragma unroll 8
      for (int c = 0; c < 32; ++c) {
        float sv = (kb + c <= qg) ? sf[r][c] : -3.0e38f;
        sf[r][c] = sv;
        rm = fmaxf(rm, sv);
      }
      float mn    = fmaxf(mo, rm);
      float scale = __expf(mo - mn);
      float lsum  = 0.f;
#pragma unroll 8
      for (int c = 0; c < 32; ++c) {
        float p = __expf(sf[r][c] - mn);
        pbuf[r][c] = (bf16)p;
        lsum += p;
      }
      mst[r] = mn;
      lst[r] = lst[r] * scale + lsum;
      scs[r] = scale;
    }
    __syncthreads();

    // Rescale running output accumulators by exp(m_old - m_new) per row.
    float sv8[8];
#pragma unroll
    for (int i = 0; i < 8; ++i) sv8[i] = scs[rb + i];
#pragma unroll
    for (int nt = 0; nt < 4; ++nt)
#pragma unroll
      for (int i = 0; i < 8; ++i) acc[nt][i] *= sv8[i];

    // O[16x64] += P[16x32] * V[kb..kb+31, :]   (B[k][n]=VT[n][kb+k])
    v16bf pa = load_a(&pbuf[0][0], 32);   // ds_load_b128 x2 per lane
#pragma unroll
    for (int nt = 0; nt < 4; ++nt)
      acc[nt] = wmma_bf16(
          pa, load_bT(VTh + (size_t)(nt * 16) * T_SEQ + kb, T_SEQ), acc[nt]);
    __syncthreads();
  }

  // Normalize by running sum and store to concat-heads buffer [T, 768].
  float inv8[8];
#pragma unroll
  for (int i = 0; i < 8; ++i) inv8[i] = 1.0f / lst[rb + i];
#pragma unroll
  for (int nt = 0; nt < 4; ++nt)
#pragma unroll
    for (int i = 0; i < 8; ++i)
      H[(size_t)(q0 + rb + i) * DM + h * DH + nt * 16 + col] =
          (bf16)(acc[nt][i] * inv8[i]);
}

// ------------------------------ output projection --------------------------
// grid = (T/16, DM/64), block = 32.  One wave -> 16x64 strip.
// out = heads @ wo + bo (fp32 out), woT is [O][C] (K-contiguous).
__global__ void __launch_bounds__(32)
out_proj_kernel(const bf16* __restrict__ hb, const bf16* __restrict__ woT,
                const float* __restrict__ bo, float* __restrict__ out) {
  const int t0 = blockIdx.x * 16;
  const int n0 = blockIdx.y * 64;

  v8f acc[4];
#pragma unroll
  for (int i = 0; i < 4; ++i) acc[i] = (v8f){0.f,0.f,0.f,0.f,0.f,0.f,0.f,0.f};

  for (int kc = 0; kc < DM; kc += 32) {
    v16bf a = load_a(hb + (size_t)t0 * DM + kc, DM);
#pragma unroll
    for (int nt = 0; nt < 4; ++nt)
      acc[nt] = wmma_bf16(
          a, load_bT(woT + (size_t)(n0 + nt * 16) * DM + kc, DM), acc[nt]);
  }

  const int lane = threadIdx.x & 31;
  const int col  = lane & 15;
  const int rb   = (lane >> 4) * 8;
#pragma unroll
  for (int nt = 0; nt < 4; ++nt) {
    const float bval = bo[n0 + nt * 16 + col];
#pragma unroll
    for (int i = 0; i < 8; ++i)
      out[(size_t)(t0 + rb + i) * DM + n0 + nt * 16 + col] = acc[nt][i] + bval;
  }
}

// ---------------------------------------------------------------------------
extern "C" void kernel_launch(void* const* d_in, const int* in_sizes, int n_in,
                              void* d_out, int out_size, void* d_ws,
                              size_t ws_size, hipStream_t stream) {
  (void)in_sizes; (void)n_in; (void)out_size; (void)ws_size;

  const float* x  = (const float*)d_in[0];
  const float* wq = (const float*)d_in[1];
  const float* bq = (const float*)d_in[2];
  const float* wk = (const float*)d_in[3];
  const float* bk = (const float*)d_in[4];
  const float* wv = (const float*)d_in[5];
  const float* bv = (const float*)d_in[6];
  const float* wo = (const float*)d_in[7];
  const float* bo = (const float*)d_in[8];

  // Workspace layout (bf16, byte offsets); total ~36.2 MB.
  char* w = (char*)d_ws;
  bf16* xb   = (bf16*)(w + 0);          // x        : 4096*768  = 6291456 B
  bf16* wqT  = (bf16*)(w + 6291456);    // wq^T     : 12*64*768 = 1179648 B
  bf16* wkT  = (bf16*)(w + 7471104);
  bf16* wvT  = (bf16*)(w + 8650752);
  bf16* woT  = (bf16*)(w + 9830400);    // wo^T     : 768*768   = 1179648 B
  bf16* qbuf = (bf16*)(w + 11010048);   // Q [H][T][Dh]         = 6291456 B
  bf16* kbuf = (bf16*)(w + 17301504);   // K [H][T][Dh]
  bf16* vTb  = (bf16*)(w + 23592960);   // V^T [H][Dh][T]
  bf16* hbuf = (bf16*)(w + 29884416);   // heads [T][768]       = 6291456 B

  const int NX = T_SEQ * DM;       // 3145728
  const int NW = NH * DM * DH;     // 589824
  const int NO = DM * DM;          // 589824

  cvt_bf16_kernel  <<<(NX + 255) / 256, 256, 0, stream>>>(x,  xb,  NX);
  cvt_wqkv_T_kernel<<<(NW + 255) / 256, 256, 0, stream>>>(wq, wqT, NW);
  cvt_wqkv_T_kernel<<<(NW + 255) / 256, 256, 0, stream>>>(wk, wkT, NW);
  cvt_wqkv_T_kernel<<<(NW + 255) / 256, 256, 0, stream>>>(wv, wvT, NW);
  cvt_wo_T_kernel  <<<(NO + 255) / 256, 256, 0, stream>>>(wo, woT, NO);

  qkv_proj_kernel<<<dim3(T_SEQ / 16, 3 * NH), 32, 0, stream>>>(
      xb, wqT, wkT, wvT, bq, bk, bv, qbuf, kbuf, vTb);

  attn_kernel<<<dim3(T_SEQ / 16, NH), 32, 0, stream>>>(qbuf, kbuf, vTb, hbuf);

  out_proj_kernel<<<dim3(T_SEQ / 16, DM / 64), 32, 0, stream>>>(
      hbuf, woT, bo, (float*)d_out);
}
